// Poly2d_61675730370816
// MI455X (gfx1250) — matrix operand order repaired and checked by reading, not compiled, not run on hardware
//
#include <hip/hip_runtime.h>
#include <hip/hip_bf16.h>

// ---------------------------------------------------------------------------
// Poly2D quadratic conv as WMMA GEMM (gfx1250 / CDNA5, wave32).
//   out[b,o,h,w] = sum_c s^T F[o,c] s + bias[o],  s = [1, 3x3 patch]
// Symmetrized: 55 unique pairs (i<=j) padded to 64 per channel.
//   A = symmetrized filters in fragment order (pre-kernel), B = on-the-fly
//   f16 pair products (16 pixels/wave), f32 accumulate.
// Half-channel software pipeline: A fragments are prefetched one k-chunk
// (4 fragments) ahead, keeping live A regs at 64 VGPRs so the allocation
// stays under 256 (no s_set_vgpr_msb churn), with sched_barrier fences so
// prefetch loads cannot sink to their consuming WMMAs.
// ---------------------------------------------------------------------------

typedef _Float16 v16h   __attribute__((ext_vector_type(16)));
typedef float    v8f    __attribute__((ext_vector_type(8)));
typedef __fp16   fp16x2 __attribute__((ext_vector_type(2)));

#define B_      4
#define C_      64
#define H_      64
#define W_      64
#define O_      64
#define HW_     (H_ * W_)
#define NPAIR   55
#define FRAG_H  512            // halves per 32x16 A fragment
#define NFRAG   (C_ * 2 * 4)   // 64 ch * 2 k-chunks * 4 cout tiles
#define APRE_HALVES (NFRAG * FRAG_H)   // 262144 halves = 512 KB

__device__ __forceinline__ unsigned pk16(float a, float b) {
    fp16x2 r = __builtin_amdgcn_cvt_pkrtz(a, b);   // v_cvt_pk_rtz_f16_f32
    return __builtin_bit_cast(unsigned, r);
}

// pair index for (i<=j), row-major over upper triangle of 10x10
#define PIDX(i, j) ((i) * 10 - ((i) * ((i) - 1)) / 2 + ((j) - (i)))

// ---------------------------------------------------------------------------
// Pre-kernel: symmetrize filters into exact A-fragment register order.
// Fragment id f = c*8 + q*4 + t ; lane l (0..31), half v (0..15).
// 16-bit A 16x32 layout: M = l%16, K = v + (v>=8 ? 8 : 0) + 8*(l/16).
// ---------------------------------------------------------------------------
__global__ void __launch_bounds__(256) poly2d_prep(const float* __restrict__ F,
                                                   _Float16* __restrict__ apre) {
    int id = blockIdx.x * blockDim.x + threadIdx.x;     // 0 .. 262143
    int v  = id & 15;
    int l  = (id >> 4) & 31;
    int t  = (id >> 9) & 3;
    int q  = (id >> 11) & 1;
    int c  = id >> 12;                                  // 0 .. 63
    int hi = l >> 4;
    int m  = l & 15;
    int k  = v + ((v >= 8) ? 8 : 0) + hi * 8;
    int p  = q * 32 + k;                                // padded pair index
    float val = 0.0f;
    if (p < NPAIR) {
        int i = 0, pr = p;
        while (pr >= (10 - i)) { pr -= (10 - i); ++i; }
        int j = i + pr;
        int o = t * 16 + m;
        const float* fb = F + ((size_t)(o * C_ + c)) * 100;
        val = fb[i * 10 + j];
        if (i != j) val += fb[j * 10 + i];              // symmetrize
    }
    apre[id] = (_Float16)val;
}

#define WMMA(ACC, A, BF)                                                      \
    ACC = __builtin_amdgcn_wmma_f32_16x16x32_f16(false, (A), false, (BF),     \
                                                 (short)0, (ACC), false, false)

// ---------------------------------------------------------------------------
// One channel: prefetch q1 frags of c, compute products(c), WMMA q0 (frags
// loaded in previous step), prefetch q0 frags of cn + patch(cn), WMMA q1.
// ---------------------------------------------------------------------------
__device__ __forceinline__ void poly2d_step(
        const _Float16* __restrict__ apre, const float* __restrict__ xb,
        int c, int cn, int lane, int hi,
        const int (&off)[9], const bool (&okm)[9],
        const float (&s)[10], float (&sn)[10],
        v16h (&aq0)[4], v16h (&aq1)[4], v8f (&acc)[4]) {
    const v16h* fr_c  = (const v16h*)(apre + (size_t)c  * (8 * FRAG_H)) + lane;
    const v16h* fr_cn = (const v16h*)(apre + (size_t)cn * (8 * FRAG_H)) + lane;

    // ---- prefetch current channel's q1 fragments (8x b128) ----
    #pragma unroll
    for (int n = 0; n < 4; ++n) aq1[n] = fr_c[(4 + n) * 32];
    __builtin_amdgcn_sched_barrier(0);

    // ---- 55 unique pair products -> 32 packed f16 pairs ----
    float pr[64];
    #pragma unroll
    for (int p = NPAIR; p < 64; ++p) pr[p] = 0.0f;
    #pragma unroll
    for (int i = 0; i < 10; ++i) {
        #pragma unroll
        for (int j = i; j < 10; ++j) pr[PIDX(i, j)] = s[i] * s[j];
    }
    unsigned pp[32];
    #pragma unroll
    for (int m = 0; m < 32; ++m) pp[m] = pk16(pr[2 * m], pr[2 * m + 1]);

    // ---- B fragments: lane's pixel, table-order halves (matches A) ----
    union { v16h h16; unsigned u[8]; } bf0, bf1;
    #pragma unroll
    for (int n = 0; n < 8; ++n) {
        const int nb = n + ((n >= 4) ? 4 : 0);
        bf0.u[n] = hi ? pp[4 + nb]  : pp[nb];        // pairs 0..31
        bf1.u[n] = hi ? pp[20 + nb] : pp[16 + nb];   // pairs 32..63
    }

    // ---- WMMA half 1 (q0 fragments, resident from previous step) ----
    WMMA(acc[0], aq0[0], bf0.h16);
    WMMA(acc[1], aq0[1], bf0.h16);
    WMMA(acc[2], aq0[2], bf0.h16);
    WMMA(acc[3], aq0[3], bf0.h16);

    // ---- prefetch next channel's q0 fragments + 3x3 patch ----
    #pragma unroll
    for (int n = 0; n < 4; ++n) aq0[n] = fr_cn[n * 32];
    const float* xc = xb + cn * HW_;
    #pragma unroll
    for (int k = 0; k < 9; ++k) {
        float vv = xc[off[k]];
        sn[1 + k] = okm[k] ? vv : 0.0f;
    }
    __builtin_amdgcn_sched_barrier(0);

    // ---- WMMA half 2 (q1 fragments, loaded at top of this step) ----
    WMMA(acc[0], aq1[0], bf1.h16);
    WMMA(acc[1], aq1[1], bf1.h16);
    WMMA(acc[2], aq1[2], bf1.h16);
    WMMA(acc[3], aq1[3], bf1.h16);
}

// ---------------------------------------------------------------------------
// Main kernel: 4 waves / block, each wave = 16 consecutive pixels of one row.
// ---------------------------------------------------------------------------
__global__ void __launch_bounds__(128)
__attribute__((amdgpu_waves_per_eu(1)))
poly2d_main(const float* __restrict__ x, const _Float16* __restrict__ apre,
            const float* __restrict__ biases, float* __restrict__ out) {
    const int lane = threadIdx.x & 31;
    const int wave = threadIdx.x >> 5;
    const int hi   = lane >> 4;          // K-half selector (wave32 layout)
    const int pl   = lane & 15;          // pixel within wave (N index)
    const int row  = blockIdx.x;         // 0 .. B*H-1
    const int b    = row >> 6;
    const int h    = row & 63;
    const int w    = wave * 16 + pl;

    v8f acc[4];
    acc[0] = {}; acc[1] = {}; acc[2] = {}; acc[3] = {};

    const float* xb = x + (size_t)(b * C_) * HW_;

    // ---- hoisted per-pixel patch addressing (clamped + validity) ----
    int  off[9];
    bool okm[9];
    #pragma unroll
    for (int dy = 0; dy < 3; ++dy) {
        int  hh  = h + dy - 1;
        int  hc  = hh < 0 ? 0 : (hh > 63 ? 63 : hh);
        bool hok = (unsigned)hh < 64u;
        #pragma unroll
        for (int dx = 0; dx < 3; ++dx) {
            int ww = w + dx - 1;
            int wc = ww < 0 ? 0 : (ww > 63 ? 63 : ww);
            off[dy * 3 + dx] = hc * W_ + wc;
            okm[dy * 3 + dx] = hok && ((unsigned)ww < 64u);
        }
    }

    // ---- prologue: preload channel 0's patch and q0 fragments ----
    float sA[10], sB[10];
    sA[0] = 1.0f; sB[0] = 1.0f;
    #pragma unroll
    for (int k = 0; k < 9; ++k) {
        float vv = xb[off[k]];
        sA[1 + k] = okm[k] ? vv : 0.0f;
    }
    v16h aq0[4], aq1[4];
    {
        const v16h* af = (const v16h*)apre + lane;     // fragment = 32 v16h
        #pragma unroll
        for (int n = 0; n < 4; ++n) aq0[n] = af[n * 32];
    }

    // ---- pipelined channel loop, ping-pong patch buffers ----
    for (int c = 0; c < C_; c += 2) {
        poly2d_step(apre, xb, c,     c + 1,        lane, hi, off, okm, sA, sB, aq0, aq1, acc);
        poly2d_step(apre, xb, c + 1, (c + 2) & 63, lane, hi, off, okm, sB, sA, aq0, aq1, acc);
    }

    // ---- epilogue: C/D layout M=cout in VGPR rows, N=pixel in lanes ----
    float* ob = out + (size_t)(b * O_) * HW_ + h * W_ + w;
    #pragma unroll
    for (int t = 0; t < 4; ++t) {
        #pragma unroll
        for (int r = 0; r < 8; ++r) {
            int o = t * 16 + r + hi * 8;
            ob[(size_t)o * HW_] = acc[t][r] + biases[o];
        }
    }
}

extern "C" void kernel_launch(void* const* d_in, const int* in_sizes, int n_in,
                              void* d_out, int out_size, void* d_ws, size_t ws_size,
                              hipStream_t stream) {
    const float* x       = (const float*)d_in[0];   // [4,64,64,64]
    const float* filters = (const float*)d_in[1];   // [64,64,10,10]
    const float* biases  = (const float*)d_in[2];   // [64,1]
    float*       out     = (float*)d_out;           // [4,64,64,64]
    _Float16*    apre    = (_Float16*)d_ws;         // 512 KB fragment-order A

    poly2d_prep<<<APRE_HALVES / 256, 256, 0, stream>>>(filters, apre);
    poly2d_main<<<B_ * H_, 128, 0, stream>>>(x, apre, biases, out);
}
